// Conditioned_User_State_Model_60584808677539
// MI455X (gfx1250) — matrix operand order repaired and checked by reading, not compiled, not run on hardware
//
#include <hip/hip_runtime.h>
#include <cstdint>
#include <cstddef>

// ---------------------------------------------------------------------------
// Problem constants (from reference): B=8192, S=256, U=128, Din=384, W=512.
// 20 ode_func evals total: 19 with dt=0.05 (floor(1/0.05)==19 in fp), then one
// final UNscaled add (reference quirk: derivative added without rest_interval).
// ---------------------------------------------------------------------------
#define B_TOT    8192
#define S_DIM    256
#define U_DIM    128
#define DIN      384
#define HID      512
#define KSTORE   512         // all weight panels stored with K padded to 512
#define MTILE    64          // batch rows per block
#define NTHREADS 256         // 8 waves (wave32)

// LDS strides (elements), padded to avoid 64-bank conflicts on 16-row /
// 16-col strided fragment accesses. PSTR = 512 data halves + 8 pad halves,
// which is exactly what the TDM pad_interval=256dw / pad_amount=4dw produces.
#define ASTR 520             // activation row stride (bf16)
#define PSTR 520             // weight-panel column stride (bf16)
#define SSTR 264             // state row stride (f32)
#define USTR 136             // user row stride (bf16)

typedef __attribute__((ext_vector_type(16))) __bf16   v16bf;
typedef __attribute__((ext_vector_type(8)))  float    v8f;
typedef __attribute__((ext_vector_type(4)))  uint32_t u32x4;
typedef __attribute__((ext_vector_type(8)))  int      i32x8;
typedef __attribute__((ext_vector_type(4)))  int      i32x4;

// ---------------------------------------------------------------------------
// WMMA fragment loaders (CDNA5 wave32 layouts, CDNA5 ISA §7.12.2)
// ---------------------------------------------------------------------------
__device__ __forceinline__ v16bf ld_a_bf16(const __bf16* buf, int stride,
                                           int rowbase, int k0, int lane) {
  const __bf16* p = buf + (rowbase + (lane & 15)) * stride + k0 + ((lane >> 4) << 3);
  union { uint4 q[2]; v16bf v; } f;
  f.q[0] = *(const uint4*)p;
  f.q[1] = *(const uint4*)(p + 16);
  return f.v;
}

__device__ __forceinline__ v16bf ld_b(const __bf16* panel, int c0, int k0, int lane) {
  const __bf16* p = panel + (c0 + (lane & 15)) * PSTR + k0 + ((lane >> 4) << 3);
  union { uint4 q[2]; v16bf v; } f;
  f.q[0] = *(const uint4*)p;
  f.q[1] = *(const uint4*)(p + 16);
  return f.v;
}

__device__ __forceinline__ v8f wmma_bf16(v16bf a, v16bf b, v8f c) {
  return __builtin_amdgcn_wmma_f32_16x16x32_bf16(false, a, false, b,
                                                 (short)0, c, false, false);
}

__device__ __forceinline__ float silu(float x) {
  return x * __builtin_amdgcn_rcpf(1.0f + __expf(-x));
}

__device__ __forceinline__ void wait_asynccnt0() {
  asm volatile("s_wait_asynccnt 0x0" ::: "memory");
}

// ---------------------------------------------------------------------------
// TDM: one tensor_load_to_lds DMAs a full 32-col x 512-half bf16 weight panel
// into LDS, inserting 4 pad dwords after every 256 data dwords (i.e. per
// column) so the LDS image lands directly in the PSTR=520 bank-padded layout.
// D# packing per CDNA5 ISA (2-D tile, groups 2/3 = 0).
// Issued by wave 0 only (TDM ignores EXEC; descriptor is wave-uniform).
// This toolchain exposes the 6-arg builtin (g0, g1, g2, g3, extra, cpol).
// ---------------------------------------------------------------------------
__device__ __forceinline__ void tdm_stage_panel(const __bf16* gsrc, uint32_t lds_off) {
  const uint64_t ga = (uint64_t)(size_t)gsrc;
  u32x4 g0;
  g0[0] = 1u;                                             // count=1 (valid), user D#
  g0[1] = lds_off;                                        // lds_addr (bytes)
  g0[2] = (uint32_t)ga;                                   // global_addr[31:0]
  g0[3] = (uint32_t)((ga >> 32) & 0x1FFFFFFu) | (2u << 30); // addr[56:32] | type=2
  i32x8 g1;
  g1[0] = (1 << 16)      // data_size = 2 bytes
        | (1 << 20)      // pad_enable
        | (7 << 22)      // pad_interval: 256 dwords (one column of 512 halves)
        | (3 << 25);     // pad_amount: 4 dwords (8 halves) -> column stride 520
  g1[1] = (uint32_t)KSTORE << 16;                         // tensor_dim0 = 512 (lo16)
  g1[2] = (32u << 16);                                    // dim0 hi16=0 | tensor_dim1 = 32
  g1[3] = ((uint32_t)KSTORE << 16);                       // dim1 hi16=0 | tile_dim0 = 512
  g1[4] = 32;                                             // tile_dim1 = 32, tile_dim2 = 0
  g1[5] = KSTORE;                                         // tensor_dim0_stride = 512 (lo32)
  g1[6] = 0;                                              // stride0 hi16 | stride1 lo16
  g1[7] = 0;                                              // stride1 hi32
  const i32x4 z4 = {0, 0, 0, 0};
  const i32x8 z8 = {0, 0, 0, 0, 0, 0, 0, 0};
  __builtin_amdgcn_tensor_load_to_lds(g0, g1, z4, z4, z8, 0);
}

__device__ __forceinline__ void wait_tensorcnt0() {
  __builtin_amdgcn_s_wait_tensorcnt((short)0);
}

__device__ __forceinline__ void prefetch_panel(const __bf16* __restrict__ WT,
                                               int c0, int tid) {
  // global_prefetch_b8 — warm L2/WGP$ for the panel-after-next
  __builtin_prefetch(WT + (size_t)(c0 + (tid >> 3)) * KSTORE + (tid & 7) * 64, 0, 1);
}

// ---------------------------------------------------------------------------
// Prep kernels: weights -> bf16, transposed to [N][K], K padded to 512.
// ---------------------------------------------------------------------------
__global__ void transpose_pad_bf16(const float* __restrict__ W, __bf16* __restrict__ WT,
                                   int K, int N) {
  int i = blockIdx.x * blockDim.x + threadIdx.x;          // over N * KSTORE
  if (i >= N * KSTORE) return;
  int n = i / KSTORE, k = i - n * KSTORE;
  WT[i] = (k < K) ? (__bf16)W[(size_t)k * N + n] : (__bf16)0.0f;
}

// ---------------------------------------------------------------------------
// Fused 20-step Euler integrator. One block owns 64 batch rows; fp32 state,
// bf16 activations and the double-buffered TDM weight panels all live in LDS
// for the entire integration. 8 waves = 4 row-groups x 2 column-halves.
// sAct2 doubles as the per-step bf16 snapshot of the state during layer 1.
// ---------------------------------------------------------------------------
__global__ __launch_bounds__(NTHREADS, 1) void ode_fused(
    const float* __restrict__ state_in, const float* __restrict__ user,
    const __bf16* __restrict__ W1T, const __bf16* __restrict__ W2T,
    const __bf16* __restrict__ W3T,
    const float* __restrict__ b1, const float* __restrict__ b2,
    const float* __restrict__ b3, float* __restrict__ out) {
  __shared__ __align__(16) float  sState[MTILE * SSTR];   // 67.6 KB fp32 master state
  __shared__ __align__(16) __bf16 sUser[MTILE * USTR];    // 17.4 KB
  __shared__ __align__(16) __bf16 sAct1[MTILE * ASTR];    // 66.6 KB
  __shared__ __align__(16) __bf16 sAct2[MTILE * ASTR];    // 66.6 KB (+ bf16 state view)
  __shared__ __align__(16) __bf16 sPanelA[32 * PSTR];     // 33.3 KB
  __shared__ __align__(16) __bf16 sPanelB[32 * PSTR];     // 33.3 KB
  __shared__ float sB1[HID], sB2[HID], sB3[S_DIM];

  const int tid  = threadIdx.x;
  const int lane = tid & 31;
  const int wave = tid >> 5;
  const int rg   = wave >> 1;          // row-group 0..3 (16 rows each)
  const int nh   = wave & 1;           // column half of the 32-col panel
  const int rowbase = rg * 16;
  const int row0 = blockIdx.x * MTILE;
  const uint32_t ldsPanelA = (uint32_t)(size_t)sPanelA;
  const uint32_t ldsPanelB = (uint32_t)(size_t)sPanelB;

  // ---- state rows: pure fp32 copy via async memory->LDS DMA (ASYNCcnt) ----
  {
    const int r = tid >> 2, seg = tid & 3;                // 64 floats per thread
    const uint64_t src = (uint64_t)(size_t)(state_in + (size_t)(row0 + r) * S_DIM + seg * 64);
    const uint32_t dst = (uint32_t)(size_t)(sState + r * SSTR + seg * 64);
#pragma unroll
    for (int i = 0; i < 16; ++i)
      asm volatile("global_load_async_to_lds_b128 %0, %1, off"
                   :: "v"(dst + 16u * i), "v"(src + 16ull * i) : "memory");
    // user params need fp32 -> bf16 conversion: plain loads + LDS stores
    const float* us = user + (size_t)(row0 + r) * U_DIM + seg * 32;
    __bf16* ud = sUser + r * USTR + seg * 32;
#pragma unroll
    for (int i = 0; i < 32; ++i) ud[i] = (__bf16)us[i];
  }
  if (tid < HID)  { sB1[tid] = b1[tid]; sB2[tid] = b2[tid]; }
  if (tid < S_DIM)  sB3[tid] = b3[tid];
  wait_asynccnt0();
  __syncthreads();

#pragma unroll 1
  for (int step = 0; step < 20; ++step) {
    const float dt = (step < 19) ? 0.05f : 1.0f;   // final eval added unscaled

    // ---- one-shot fp32 -> bf16 state snapshot into sAct2 (dead in layer 1) ----
    {
      const int r = tid >> 2, seg = tid & 3;
      const float4* sp = (const float4*)(sState + r * SSTR + seg * 64);
      uint4* dp = (uint4*)(sAct2 + r * ASTR + seg * 64);
#pragma unroll
      for (int i = 0; i < 8; ++i) {                 // 8 x (8 floats -> 8 bf16)
        float4 a = sp[2 * i], b = sp[2 * i + 1];
        union { __bf16 h[8]; uint4 q; } u;
        u.h[0] = (__bf16)a.x; u.h[1] = (__bf16)a.y;
        u.h[2] = (__bf16)a.z; u.h[3] = (__bf16)a.w;
        u.h[4] = (__bf16)b.x; u.h[5] = (__bf16)b.y;
        u.h[6] = (__bf16)b.z; u.h[7] = (__bf16)b.w;
        dp[i] = u.q;
      }
    }
    if (wave == 0) { tdm_stage_panel(W1T, ldsPanelA); wait_tensorcnt0(); }
    __syncthreads();

    // ======== layer 1: [state|user] (64x384) @ W1 (384x512) -> act1 ========
#pragma unroll 1
    for (int p = 0; p < HID / 32; ++p) {
      const __bf16* cur = (p & 1) ? sPanelB : sPanelA;
      const uint32_t nxt = (p & 1) ? ldsPanelA : ldsPanelB;
      if (wave == 0 && p + 1 < HID / 32)
        tdm_stage_panel(W1T + (size_t)(p + 1) * 32 * KSTORE, nxt);
      if (p + 2 < HID / 32) prefetch_panel(W1T, (p + 2) * 32, tid);
      v8f c = {};
#pragma unroll
      for (int k0 = 0; k0 < S_DIM; k0 += 32)       // bf16 state snapshot
        c = wmma_bf16(ld_a_bf16(sAct2, ASTR, rowbase, k0, lane),
                      ld_b(cur, nh * 16, k0, lane), c);
#pragma unroll
      for (int k0 = 0; k0 < U_DIM; k0 += 32)       // user part
        c = wmma_bf16(ld_a_bf16(sUser, USTR, rowbase, k0, lane),
                      ld_b(cur, nh * 16, S_DIM + k0, lane), c);
      const int col  = p * 32 + nh * 16 + (lane & 15);
      const int rr   = rowbase + ((lane >> 4) << 3);
      const float bb = sB1[col];
#pragma unroll
      for (int v = 0; v < 8; ++v)
        sAct1[(rr + v) * ASTR + col] = (__bf16)silu(c[v] + bb);
      wait_tensorcnt0();
      __syncthreads();
    }

    // ======== layer 2: act1 (64x512) @ W2 (512x512) -> act2 ========
    if (wave == 0) { tdm_stage_panel(W2T, ldsPanelA); wait_tensorcnt0(); }
    __syncthreads();
#pragma unroll 1
    for (int p = 0; p < HID / 32; ++p) {
      const __bf16* cur = (p & 1) ? sPanelB : sPanelA;
      const uint32_t nxt = (p & 1) ? ldsPanelA : ldsPanelB;
      if (wave == 0 && p + 1 < HID / 32)
        tdm_stage_panel(W2T + (size_t)(p + 1) * 32 * KSTORE, nxt);
      if (p + 2 < HID / 32) prefetch_panel(W2T, (p + 2) * 32, tid);
      v8f c = {};
#pragma unroll
      for (int k0 = 0; k0 < HID; k0 += 32)
        c = wmma_bf16(ld_a_bf16(sAct1, ASTR, rowbase, k0, lane),
                      ld_b(cur, nh * 16, k0, lane), c);
      const int col  = p * 32 + nh * 16 + (lane & 15);
      const int rr   = rowbase + ((lane >> 4) << 3);
      const float bb = sB2[col];
#pragma unroll
      for (int v = 0; v < 8; ++v)
        sAct2[(rr + v) * ASTR + col] = (__bf16)silu(c[v] + bb);
      wait_tensorcnt0();
      __syncthreads();
    }

    // ======== layer 3: act2 (64x512) @ W3 (512x256) -> state += dt*(. + b3) ==
    if (wave == 0) { tdm_stage_panel(W3T, ldsPanelA); wait_tensorcnt0(); }
    __syncthreads();
#pragma unroll 1
    for (int p = 0; p < S_DIM / 32; ++p) {
      const __bf16* cur = (p & 1) ? sPanelB : sPanelA;
      const uint32_t nxt = (p & 1) ? ldsPanelA : ldsPanelB;
      if (wave == 0 && p + 1 < S_DIM / 32)
        tdm_stage_panel(W3T + (size_t)(p + 1) * 32 * KSTORE, nxt);
      if (p + 2 < S_DIM / 32) prefetch_panel(W3T, (p + 2) * 32, tid);
      v8f c = {};
#pragma unroll
      for (int k0 = 0; k0 < HID; k0 += 32)
        c = wmma_bf16(ld_a_bf16(sAct2, ASTR, rowbase, k0, lane),
                      ld_b(cur, nh * 16, k0, lane), c);
      const int col  = p * 32 + nh * 16 + (lane & 15);
      const int rr   = rowbase + ((lane >> 4) << 3);
      const float bb = sB3[col];
#pragma unroll
      for (int v = 0; v < 8; ++v)
        sState[(rr + v) * SSTR + col] += dt * (c[v] + bb);   // disjoint per lane
      wait_tensorcnt0();
      __syncthreads();
    }
  }

  // ---- write final fp32 state ----
  {
    const int r = tid >> 2, seg = tid & 3;
    const float4* src = (const float4*)(sState + r * SSTR + seg * 64);
    float4* dst = (float4*)(out + (size_t)(row0 + r) * S_DIM + seg * 64);
#pragma unroll
    for (int i = 0; i < 16; ++i) dst[i] = src[i];
  }
}

// ---------------------------------------------------------------------------
// Inputs (setup_inputs order): state, user_params, W1, b1, W2, b2, W3, b3, h.
// Output: final state [8192, 256] fp32. Workspace: bf16 [N][Kpad=512]
// transposed weights (1.31 MB). h==1 -> 19 scaled steps + 1 unscaled.
// ---------------------------------------------------------------------------
extern "C" void kernel_launch(void* const* d_in, const int* in_sizes, int n_in,
                              void* d_out, int out_size, void* d_ws, size_t ws_size,
                              hipStream_t stream) {
  (void)in_sizes; (void)n_in; (void)out_size; (void)ws_size;
  const float* state = (const float*)d_in[0];
  const float* user  = (const float*)d_in[1];
  const float* W1    = (const float*)d_in[2];
  const float* b1    = (const float*)d_in[3];
  const float* W2    = (const float*)d_in[4];
  const float* b2    = (const float*)d_in[5];
  const float* W3    = (const float*)d_in[6];
  const float* b3    = (const float*)d_in[7];

  __bf16* W1T = (__bf16*)d_ws;            // [512][512] (K padded from 384)
  __bf16* W2T = W1T + HID * KSTORE;       // [512][512]
  __bf16* W3T = W2T + HID * KSTORE;       // [256][512]
  float* out = (float*)d_out;

  const int n1 = HID * KSTORE, n2 = HID * KSTORE, n3 = S_DIM * KSTORE;
  transpose_pad_bf16<<<(n1 + 255) / 256, 256, 0, stream>>>(W1, W1T, DIN, HID);
  transpose_pad_bf16<<<(n2 + 255) / 256, 256, 0, stream>>>(W2, W2T, HID, HID);
  transpose_pad_bf16<<<(n3 + 255) / 256, 256, 0, stream>>>(W3, W3T, HID, S_DIM);
  ode_fused<<<B_TOT / MTILE, NTHREADS, 0, stream>>>(state, user, W1T, W2T, W3T,
                                                    b1, b2, b3, out);
}
